// MultiHeadAttentionModule_38113539784745
// MI455X (gfx1250) — matrix operand order repaired and checked by reading, not compiled, not run on hardware
//
#include <hip/hip_runtime.h>
#include <hip/hip_bf16.h>
#include <math.h>

typedef __attribute__((ext_vector_type(16))) _Float16 v16h;
typedef __attribute__((ext_vector_type(8)))  float    v8f;

constexpr int D_MODEL = 2048;
constexpr int NHEADS  = 16;
constexpr int DK      = 128;
constexpr int BATCH   = 2;
constexpr int SEQ     = 2048;
constexpr int MTOK    = BATCH * SEQ;          // 4096 token rows

struct alignas(16) U4 { unsigned x, y, z, w; };
struct alignas(16) F4 { float a, b, c, d; };

union Frag {
    v16h     h;
    U4       q[2];
    unsigned u[8];
};

union Pack8 {            // 8 contiguous f16 = one 16-byte store
    _Float16 h[8];
    U4       q;
};

__device__ inline v8f wmma16(const Frag& a, const Frag& b, v8f c) {
    // emits v_wmma_f32_16x16x32_f16
    return __builtin_amdgcn_wmma_f32_16x16x32_f16(
        /*neg_a=*/false, a.h, /*neg_b=*/false, b.h,
        /*c_mod=*/(short)0, c, /*reuse_a=*/false, /*reuse_b=*/false);
}

// ---------------------------------------------------------------------------
// f32 -> f16 conversion, 8 elements / thread (2x b128 in, 1x b128 out)
// ---------------------------------------------------------------------------
__global__ void mha_cvt_f16_v8(const float* __restrict__ src,
                               _Float16* __restrict__ dst, size_t n8) {
    size_t i = (size_t)blockIdx.x * blockDim.x + threadIdx.x;
    if (i >= n8) return;
    const F4 f0 = ((const F4*)src)[2 * i];
    const F4 f1 = ((const F4*)src)[2 * i + 1];
    Pack8 pk;
    pk.h[0] = (_Float16)f0.a; pk.h[1] = (_Float16)f0.b;
    pk.h[2] = (_Float16)f0.c; pk.h[3] = (_Float16)f0.d;
    pk.h[4] = (_Float16)f1.a; pk.h[5] = (_Float16)f1.b;
    pk.h[6] = (_Float16)f1.c; pk.h[7] = (_Float16)f1.d;
    ((U4*)dst)[i] = pk.q;
}

// ---------------------------------------------------------------------------
// WMMA GEMM:  C[m,n] = sum_k A[m,k] * W[n,k]
// Block = 256 threads = 8 waves. Each wave: 32x64 tile (2 M-tiles x 4 N-tiles,
// 8 accumulators), software-pipelined K loop (double-buffered fragments).
//   mode 0: f16 out, layout [B,H,S,DK]   (Q,K pre-RoPE)
//   mode 1: f16 out, layout [B,H,DK,S]   (V transposed; packed b128 stores)
//   mode 2: f32 out, layout [MTOK, D_MODEL]  (final projection -> d_out)
// ---------------------------------------------------------------------------
__global__ __launch_bounds__(256) void mha_gemm_wmma(
    const _Float16* __restrict__ A, const _Float16* __restrict__ W,
    void* __restrict__ dstv, int mode)
{
    const int lane = threadIdx.x & 31;
    const int wave = threadIdx.x >> 5;
    const int half = lane >> 4;       // 0: lanes 0-15, 1: lanes 16-31
    const int nl   = lane & 15;
    const int m0   = blockIdx.y * 32;
    const int n0   = blockIdx.x * 512 + wave * 64;
    const int koffA = half * 16;      // A-frag: contiguous 16 K per lane half
    const int koffB = half * 16;      // B-frag: contiguous 16 K per lane half

    const _Float16* arow[2] = { A + (size_t)(m0 + nl) * D_MODEL,
                                A + (size_t)(m0 + 16 + nl) * D_MODEL };
    const _Float16* wrow[4];
#pragma unroll
    for (int j = 0; j < 4; ++j)
        wrow[j] = W + (size_t)(n0 + j * 16 + nl) * D_MODEL + koffB;

    const v8f Z = {0.f,0.f,0.f,0.f,0.f,0.f,0.f,0.f};
    v8f acc[2][4] = {{Z,Z,Z,Z},{Z,Z,Z,Z}};

    // ---- prologue: load fragments for kb = 0 ----
    Frag a_cur[2], b_cur[4];
#pragma unroll
    for (int i = 0; i < 2; ++i) {
        a_cur[i].q[0] = *(const U4*)(arow[i] + koffA);
        a_cur[i].q[1] = *(const U4*)(arow[i] + 8 + koffA);
    }
#pragma unroll
    for (int j = 0; j < 4; ++j) {
        b_cur[j].q[0] = *(const U4*)(wrow[j]);
        b_cur[j].q[1] = *(const U4*)(wrow[j] + 8);
    }

    // ---- pipelined main loop: load kb+32 while computing kb ----
    for (int kb = 0; kb < D_MODEL - 32; kb += 32) {
        const int kn = kb + 32;
        __builtin_prefetch(arow[0] + kn + 192, 0, 3);   // global_prefetch_b8
        Frag a_nxt[2], b_nxt[4];
#pragma unroll
        for (int i = 0; i < 2; ++i) {
            a_nxt[i].q[0] = *(const U4*)(arow[i] + kn + koffA);
            a_nxt[i].q[1] = *(const U4*)(arow[i] + kn + 8 + koffA);
        }
#pragma unroll
        for (int j = 0; j < 4; ++j) {
            b_nxt[j].q[0] = *(const U4*)(wrow[j] + kn);
            b_nxt[j].q[1] = *(const U4*)(wrow[j] + kn + 8);
        }
#pragma unroll
        for (int i = 0; i < 2; ++i)
#pragma unroll
            for (int j = 0; j < 4; ++j)
                acc[i][j] = wmma16(a_cur[i], b_cur[j], acc[i][j]);
#pragma unroll
        for (int i = 0; i < 2; ++i) a_cur[i] = a_nxt[i];
#pragma unroll
        for (int j = 0; j < 4; ++j) b_cur[j] = b_nxt[j];
    }
    // ---- epilogue K-step ----
#pragma unroll
    for (int i = 0; i < 2; ++i)
#pragma unroll
        for (int j = 0; j < 4; ++j)
            acc[i][j] = wmma16(a_cur[i], b_cur[j], acc[i][j]);

    // ---- store. C layout: vgpr r <-> row M = half*8 + r, lane%16 <-> col ----
#pragma unroll
    for (int i = 0; i < 2; ++i) {
#pragma unroll
        for (int j = 0; j < 4; ++j) {
            const int n  = n0 + j * 16 + nl;
            const int mb = m0 + i * 16 + half * 8;       // 8 contiguous rows
            if (mode == 1) {
                // vt[b,h,dk,s]: lane's 8 rows are contiguous in s -> b128 store
                const int b  = mb / SEQ, s = mb % SEQ;
                const int hh = n / DK,   dk = n % DK;
                Pack8 pk;
#pragma unroll
                for (int r = 0; r < 8; ++r) pk.h[r] = (_Float16)acc[i][j][r];
                *(U4*)((_Float16*)dstv +
                       (((size_t)(b * NHEADS + hh)) * DK + dk) * SEQ + s) = pk.q;
            } else {
#pragma unroll
                for (int r = 0; r < 8; ++r) {
                    const int m = mb + r;
                    const float val = acc[i][j][r];
                    if (mode == 2) {
                        ((float*)dstv)[(size_t)m * D_MODEL + n] = val;
                    } else {
                        const int b  = m / SEQ, s = m % SEQ;
                        const int hh = n / DK,  dk = n % DK;
                        ((_Float16*)dstv)[(((size_t)(b * NHEADS + hh)) * SEQ + s) * DK + dk]
                            = (_Float16)val;
                    }
                }
            }
        }
    }
}

// ---------------------------------------------------------------------------
// RoPE on Q and K, layout [B*H, S, DK], f16 in/out, fp32 math.
// Folds the 1/sqrt(DK) score scaling into Q.
// ---------------------------------------------------------------------------
__global__ void mha_rope(_Float16* __restrict__ qh, _Float16* __restrict__ kh,
                         const int* __restrict__ pos, float qscale)
{
    size_t idx = (size_t)blockIdx.x * blockDim.x + threadIdx.x;
    const int    i    = (int)(idx & 63);          // rotation pair index 0..63
    const size_t rest = idx >> 6;
    const int    s    = (int)(rest % SEQ);
    const size_t bh   = rest / SEQ;
    if (bh >= (size_t)BATCH * NHEADS) return;

    const float p   = (float)pos[s];
    const float inv = powf(10000.0f, -(2.0f * (float)i) / (float)DK);
    float c, sn;
    sincosf(p * inv, &c, &sn);

    const size_t base = (bh * SEQ + s) * DK + 2 * i;
    {
        float e = (float)qh[base], o = (float)qh[base + 1];
        qh[base]     = (_Float16)((e * c - o * sn) * qscale);
        qh[base + 1] = (_Float16)((e * sn + o * c) * qscale);
    }
    {
        float e = (float)kh[base], o = (float)kh[base + 1];
        kh[base]     = (_Float16)(e * c - o * sn);
        kh[base + 1] = (_Float16)(e * sn + o * c);
    }
}

// ---------------------------------------------------------------------------
// Flash attention: one wave per 16-query tile, online softmax (fp32 stats),
// 32 keys / iteration.
//   - all 8 K-fragments loaded in one clause, then 8 chained score WMMAs
//   - all 8 V-fragments issued BEFORE the softmax VALU block (latency hiding)
//   - P transposed C-layout -> A-layout via per-wave LDS + s_wait_dscnt
// ---------------------------------------------------------------------------
__global__ __launch_bounds__(128) void mha_flash_attn(
    const _Float16* __restrict__ qh, const _Float16* __restrict__ kh,
    const _Float16* __restrict__ vt, _Float16* __restrict__ ah)
{
    __shared__ _Float16 plds[4][16 * 32];   // 1KB per wave

    const int lane = threadIdx.x & 31;
    const int wave = threadIdx.x >> 5;
    const int half = lane >> 4;
    const int nl   = lane & 15;
    const int bh   = blockIdx.y;                       // 0 .. B*H-1
    const int mt   = blockIdx.x * 4 + wave;            // query tile
    const int q0   = mt * 16;

    const _Float16* Qb = qh + (size_t)bh * SEQ * DK;
    const _Float16* Kb = kh + (size_t)bh * SEQ * DK;
    const _Float16* Vb = vt + (size_t)bh * DK * SEQ;

    // Q fragments for all 4 d-chunks (kept in registers for the whole tile)
    Frag qf[4];
    const _Float16* qrow = Qb + (size_t)(q0 + nl) * DK;
#pragma unroll
    for (int c = 0; c < 4; ++c) {
        qf[c].q[0] = *(const U4*)(qrow + c * 32 + half * 16);
        qf[c].q[1] = *(const U4*)(qrow + c * 32 + 8 + half * 16);
    }

    const v8f Z = {0.f,0.f,0.f,0.f,0.f,0.f,0.f,0.f};
    v8f   o[8] = {Z, Z, Z, Z, Z, Z, Z, Z};   // 16 q x 128 d accumulators
    float mstat[8], lstat[8];
#pragma unroll
    for (int r = 0; r < 8; ++r) { mstat[r] = -INFINITY; lstat[r] = 0.f; }

    const int kp_max = (q0 + 15) / 32;
    for (int kp = 0; kp <= kp_max; ++kp) {
        const int kb = kp * 32;

        // ---- batch-load all 8 K-fragments (one clause) ----
        Frag kf[8];
#pragma unroll
        for (int c = 0; c < 4; ++c) {
            const _Float16* k0 = Kb + (size_t)(kb + nl)      * DK + c * 32 + half * 16;
            const _Float16* k1 = Kb + (size_t)(kb + 16 + nl) * DK + c * 32 + half * 16;
            kf[c].q[0]     = *(const U4*)k0;  kf[c].q[1]     = *(const U4*)(k0 + 8);
            kf[4 + c].q[0] = *(const U4*)k1;  kf[4 + c].q[1] = *(const U4*)(k1 + 8);
        }
        // ---- scores: two interleaved 4-WMMA chains over d_k = 128 ----
        v8f s0 = Z, s1 = Z;
#pragma unroll
        for (int c = 0; c < 4; ++c) {
            s0 = wmma16(qf[c], kf[c],     s0);
            s1 = wmma16(qf[c], kf[4 + c], s1);
        }

        // ---- issue all 8 V-fragment loads now; softmax VALU hides latency ----
        Frag vf[8];
#pragma unroll
        for (int c = 0; c < 8; ++c) {
            const _Float16* vp = Vb + (size_t)(c * 16 + nl) * SEQ + kb + half * 16;
            vf[c].q[0] = *(const U4*)vp;  vf[c].q[1] = *(const U4*)(vp + 8);
        }

        // ---- causal mask + online softmax (per row, across 16-lane group) --
        float p0[8], p1[8];
#pragma unroll
        for (int r = 0; r < 8; ++r) {
            const int qi = q0 + half * 8 + r;
            float v0 = (kb + nl)      <= qi ? s0[r] : -INFINITY;
            float v1 = (kb + 16 + nl) <= qi ? s1[r] : -INFINITY;
            float mx = fmaxf(v0, v1);
            mx = fmaxf(mx, __shfl_xor(mx, 1, 32));
            mx = fmaxf(mx, __shfl_xor(mx, 2, 32));
            mx = fmaxf(mx, __shfl_xor(mx, 4, 32));
            mx = fmaxf(mx, __shfl_xor(mx, 8, 32));
            const float mnew  = fmaxf(mstat[r], mx);
            const float alpha = expf(mstat[r] - mnew);
            const float e0 = expf(v0 - mnew);
            const float e1 = expf(v1 - mnew);
            float rs = e0 + e1;
            rs += __shfl_xor(rs, 1, 32);
            rs += __shfl_xor(rs, 2, 32);
            rs += __shfl_xor(rs, 4, 32);
            rs += __shfl_xor(rs, 8, 32);
            lstat[r] = lstat[r] * alpha + rs;
            mstat[r] = mnew;
            p0[r] = e0; p1[r] = e1;
#pragma unroll
            for (int c = 0; c < 8; ++c) o[c][r] *= alpha;
        }

        // ---- transpose P (C-layout) -> A-layout via per-wave LDS ----
        _Float16* pl = plds[wave];
#pragma unroll
        for (int r = 0; r < 8; ++r) {
            const int mrow = half * 8 + r;
            pl[mrow * 32 + nl]      = (_Float16)p0[r];
            pl[mrow * 32 + 16 + nl] = (_Float16)p1[r];
        }
        asm volatile("s_wait_dscnt 0" ::: "memory");
        Frag pf;
        pf.q[0] = *(const U4*)(pl + nl * 32 + half * 16);
        pf.q[1] = *(const U4*)(pl + nl * 32 + 8 + half * 16);

        // ---- P x V : 8 WMMAs over the 128 output dims ----
#pragma unroll
        for (int c = 0; c < 8; ++c)
            o[c] = wmma16(pf, vf[c], o[c]);
    }

    // ---- finalize: divide by l, write f16 to [B,S,D_MODEL] ----
    const int b = bh / NHEADS, h = bh % NHEADS;
#pragma unroll
    for (int c = 0; c < 8; ++c) {
#pragma unroll
        for (int r = 0; r < 8; ++r) {
            const int s = q0 + half * 8 + r;
            const float val = o[c][r] / lstat[r];
            ah[((size_t)(b * SEQ + s)) * D_MODEL + h * DK + c * 16 + nl] = (_Float16)val;
        }
    }
}

// ---------------------------------------------------------------------------
extern "C" void kernel_launch(void* const* d_in, const int* in_sizes, int n_in,
                              void* d_out, int out_size, void* d_ws, size_t ws_size,
                              hipStream_t stream) {
    const float* x  = (const float*)d_in[0];
    const float* wq = (const float*)d_in[1];
    const float* wk = (const float*)d_in[2];
    const float* wv = (const float*)d_in[3];
    const float* wo = (const float*)d_in[4];
    const int*  pos = (const int*)d_in[5];
    float* out = (float*)d_out;

    constexpr size_t MB = 1ull << 20;
    constexpr size_t NX = (size_t)MTOK * D_MODEL;       // 8M elements
    constexpr size_t NW = (size_t)D_MODEL * D_MODEL;    // 4M elements

    char* ws = (char*)d_ws;
    _Float16* xh  = (_Float16*)(ws +   0 * MB);   // 16 MB
    _Float16* wqh = (_Float16*)(ws +  16 * MB);   //  8 MB
    _Float16* wkh = (_Float16*)(ws +  24 * MB);   //  8 MB
    _Float16* wvh = (_Float16*)(ws +  32 * MB);   //  8 MB
    _Float16* woh = (_Float16*)(ws +  40 * MB);   //  8 MB
    _Float16* qh  = (_Float16*)(ws +  48 * MB);   // 16 MB  [B,H,S,DK]
    _Float16* kh  = (_Float16*)(ws +  64 * MB);   // 16 MB  [B,H,S,DK]
    _Float16* vt  = (_Float16*)(ws +  80 * MB);   // 16 MB  [B,H,DK,S]
    _Float16* ah  = (_Float16*)(ws +  96 * MB);   // 16 MB  [B,S,D]

    // 1) f32 -> f16 conversions (8 elems/thread, b128 in/out)
    mha_cvt_f16_v8<<<(unsigned)((NX / 8 + 255) / 256), 256, 0, stream>>>(x,  xh,  NX / 8);
    mha_cvt_f16_v8<<<(unsigned)((NW / 8 + 255) / 256), 256, 0, stream>>>(wq, wqh, NW / 8);
    mha_cvt_f16_v8<<<(unsigned)((NW / 8 + 255) / 256), 256, 0, stream>>>(wk, wkh, NW / 8);
    mha_cvt_f16_v8<<<(unsigned)((NW / 8 + 255) / 256), 256, 0, stream>>>(wv, wvh, NW / 8);
    mha_cvt_f16_v8<<<(unsigned)((NW / 8 + 255) / 256), 256, 0, stream>>>(wo, woh, NW / 8);

    // 2) QKV projections (WMMA GEMMs), 32 rows x 512 cols per block
    dim3 ggrid(D_MODEL / 512, MTOK / 32);   // (4, 128)
    mha_gemm_wmma<<<ggrid, 256, 0, stream>>>(xh, wqh, (void*)qh, 0);
    mha_gemm_wmma<<<ggrid, 256, 0, stream>>>(xh, wkh, (void*)kh, 0);
    mha_gemm_wmma<<<ggrid, 256, 0, stream>>>(xh, wvh, (void*)vt, 1);

    // 3) RoPE (scale folded into Q)
    const size_t nrope = (size_t)BATCH * NHEADS * SEQ * (DK / 2);
    mha_rope<<<(unsigned)((nrope + 255) / 256), 256, 0, stream>>>(
        qh, kh, pos, 0.08838834764831845f /* 1/sqrt(128) */);

    // 4) Flash attention (WMMA QK^T and P*V)
    mha_flash_attn<<<dim3(SEQ / 64, BATCH * NHEADS), 128, 0, stream>>>(qh, kh, vt, ah);

    // 5) Output projection straight into fp32 d_out
    mha_gemm_wmma<<<ggrid, 256, 0, stream>>>(ah, woh, (void*)out, 2);
}
